// GNNClassifier_28716151341663
// MI455X (gfx1250) — matrix operand order repaired and checked by reading, not compiled, not run on hardware
//
#include <hip/hip_runtime.h>
#include <hip/hip_bf16.h>
#include <math.h>

#define N_NODES    100000
#define N_EDGES    1600000
#define HIDDEN     128
#define NUM_GRAPHS 256
#define SCAN_BLK   256
#define N_SCAN_BLK ((N_NODES + SCAN_BLK - 1) / SCAN_BLK)   // 391
#define POOL_NODES 64

typedef __attribute__((ext_vector_type(16))) _Float16 v16h;
typedef __attribute__((ext_vector_type(8)))  _Float16 v8h;
typedef __attribute__((ext_vector_type(8)))  float    v8f;

// ---------------- degree / scalar layer-1 pipeline --------------------------------

__global__ void k_init_deg(float* __restrict__ deg) {
    int n = blockIdx.x * blockDim.x + threadIdx.x;
    if (n < N_NODES) deg[n] = 1.0f;            // self-loop
}

__global__ void k_count_deg(const int* __restrict__ col, float* __restrict__ deg) {
    int e = blockIdx.x * blockDim.x + threadIdx.x;
    if (e < N_EDGES) atomicAdd(&deg[col[e]], 1.0f);
}

__global__ void k_dinv_u(const float* __restrict__ deg, const float* __restrict__ x,
                         float* __restrict__ dinv, float* __restrict__ u,
                         float* __restrict__ t, int* __restrict__ degi) {
    int n = blockIdx.x * blockDim.x + threadIdx.x;
    if (n < N_NODES) {
        float d  = deg[n];
        float di = rsqrtf(d);                  // deg >= 1 always (self-loops)
        dinv[n] = di;
        u[n]    = x[n] * di;
        t[n]    = 0.0f;
        degi[n] = (int)(d + 0.5f) - 1;         // in-degree excluding self-loop
    }
}

__global__ void k_scatter_u(const int* __restrict__ row, const int* __restrict__ col,
                            const float* __restrict__ u, float* __restrict__ t) {
    int e = blockIdx.x * blockDim.x + threadIdx.x;
    if (e < N_EDGES) atomicAdd(&t[col[e]], u[row[e]]);
}

__global__ void k_s1(const float* __restrict__ dinv, const float* __restrict__ u,
                     const float* __restrict__ t, float* __restrict__ s1) {
    int n = blockIdx.x * blockDim.x + threadIdx.x;
    if (n < N_NODES) s1[n] = dinv[n] * (t[n] + u[n]);   // includes self-loop term
}

// ---------------- CSR build (counting sort by target node) ------------------------

__global__ void k_blocksum(const int* __restrict__ degi, int* __restrict__ blockSums) {
    __shared__ int sh[SCAN_BLK];
    int t = threadIdx.x, n = blockIdx.x * SCAN_BLK + t;
    sh[t] = (n < N_NODES) ? degi[n] : 0;
    __syncthreads();
    #pragma unroll
    for (int s = SCAN_BLK / 2; s > 0; s >>= 1) {
        if (t < s) sh[t] += sh[t + s];
        __syncthreads();
    }
    if (t == 0) blockSums[blockIdx.x] = sh[0];
}

__global__ void k_scan_blocks(const int* __restrict__ blockSums, int* __restrict__ blockOff) {
    if (blockIdx.x == 0 && threadIdx.x == 0) {
        int run = 0;
        for (int i = 0; i < N_SCAN_BLK; ++i) { blockOff[i] = run; run += blockSums[i]; }
    }
}

__global__ void k_scan_local(const int* __restrict__ degi, const int* __restrict__ blockOff,
                             int* __restrict__ rowptr, int* __restrict__ cursor) {
    __shared__ int sh[SCAN_BLK];
    int t = threadIdx.x, b = blockIdx.x, n = b * SCAN_BLK + t;
    int v = (n < N_NODES) ? degi[n] : 0;
    sh[t] = v;
    __syncthreads();
    #pragma unroll
    for (int off = 1; off < SCAN_BLK; off <<= 1) {        // Hillis-Steele inclusive scan
        int add = (t >= off) ? sh[t - off] : 0;
        __syncthreads();
        sh[t] += add;
        __syncthreads();
    }
    if (n < N_NODES) {
        int p = blockOff[b] + sh[t] - v;                  // exclusive prefix
        rowptr[n] = p;
        cursor[n] = p;
    }
}

__global__ void k_bucket(const int* __restrict__ row, const int* __restrict__ col,
                         int* __restrict__ cursor, int* __restrict__ ebuf) {
    int e = blockIdx.x * blockDim.x + threadIdx.x;
    if (e < N_EDGES) {
        int pos = atomicAdd(&cursor[col[e]], 1);
        ebuf[pos] = row[e];
    }
}

// ---------------- atomic-free layer-2 aggregation: A[c,:] = sum_e norm*h1[row] -----
// h1[n,k] = relu(s1[n]*W1[k] + b1[k]) computed on the fly from the scalar s1.
// One wave per node; lane holds 4 feature accumulators in registers.

__global__ void k_agg_csr(const int* __restrict__ rowptr, const int* __restrict__ degi,
                          const int* __restrict__ ebuf, const float* __restrict__ dinv,
                          const float* __restrict__ s1, const float* __restrict__ W1,
                          const float* __restrict__ b1, float* __restrict__ A) {
    int wv   = threadIdx.x >> 5;
    int lane = threadIdx.x & 31;
    int c    = blockIdx.x * 8 + wv;            // 100000 / 8 = 12500 blocks exact
    int k0   = lane * 4;
    float w1r[4], b1r[4];
    #pragma unroll
    for (int i = 0; i < 4; ++i) { w1r[i] = W1[k0 + i]; b1r[i] = b1[k0 + i]; }
    float dc = dinv[c], sc = s1[c];
    float acc[4];
    #pragma unroll
    for (int i = 0; i < 4; ++i) {              // self-loop term: dinv[c]^2 * h1[c]
        float h = sc * w1r[i] + b1r[i];
        acc[i] = dc * dc * (h > 0.0f ? h : 0.0f);
    }
    int start = rowptr[c], cnt = degi[c];
    for (int j = 0; j < cnt; ++j) {
        int   r  = ebuf[start + j];
        float sr = s1[r];
        float w  = dc * dinv[r];
        #pragma unroll
        for (int i = 0; i < 4; ++i) {
            float h = sr * w1r[i] + b1r[i];
            acc[i] += w * (h > 0.0f ? h : 0.0f);
        }
    }
    float4 o = make_float4(acc[0], acc[1], acc[2], acc[3]);
    *(float4*)(A + (size_t)c * HIDDEN + k0) = o;
}

// ---------------- pack W2 -> f16 column-major (B fragments load contiguously) ------

__global__ void k_pack_W2(const float* __restrict__ W2, _Float16* __restrict__ Bt) {
    int i = blockIdx.x * blockDim.x + threadIdx.x;
    if (i >= HIDDEN * HIDDEN) return;
    int k = i >> 7, c = i & 127;
    Bt[(size_t)c * HIDDEN + k] = (_Float16)W2[i];
}

// ---------------- WMMA GEMM: H2 = relu(A @ W2 + b2), [100000,128]x[128,128] --------

__global__ void k_gemm_wmma(const float* __restrict__ A, const _Float16* __restrict__ Bt,
                            const float* __restrict__ b2, float* __restrict__ H2) {
    __shared__ __align__(32) _Float16 Atile[16 * HIDDEN];   // 16x128 f16 tile (4 KB)
    const int r0 = blockIdx.x * 16;
    const int t  = threadIdx.x;                // 256 threads = 8 wave32

    {   // stage + convert 16x128 fp32 -> f16 into LDS (8 elems/thread, contiguous)
        int idx = t * 8;
        int rr  = idx >> 7;
        int cc  = idx & 127;
        const float* src = A + (size_t)(r0 + rr) * HIDDEN + cc;
        #pragma unroll
        for (int i = 0; i < 8; ++i) Atile[idx + i] = (_Float16)src[i];
    }
    __syncthreads();

    const int wave = t >> 5;
    const int lane = t & 31;
    const int c0   = wave * 16;                // this wave's 16 output columns
    const int half = lane >> 4;                // lane group 0-15 / 16-31
    const int lm   = lane & 15;

    v8f acc = {};
    #pragma unroll
    for (int kb = 0; kb < 4; ++kb) {
        // A fragment (16x32 f16): lanes0-15 K={0..15}, lanes16-31 K={16..31}
        const int k0 = kb * 32 + half * 16;
        const int k1 = k0 + 8;
        v8h a0 = *(const v8h*)&Atile[lm * HIDDEN + k0];
        v8h a1 = *(const v8h*)&Atile[lm * HIDDEN + k1];
        v16h av;
        #pragma unroll
        for (int i = 0; i < 8; ++i) { av[i] = a0[i]; av[8 + i] = a1[i]; }
        // B fragment (32x16 f16): lane = column, 16 contiguous K per lane-half
        v16h bv = *(const v16h*)(Bt + (size_t)(c0 + lm) * HIDDEN + kb * 32 + half * 16);
        acc = __builtin_amdgcn_wmma_f32_16x16x32_f16(
            false, av, false, bv, (short)0, acc, false, false);
    }

    // D layout: VGPR j -> row j (lanes 0-15) / row j+8 (lanes 16-31), col = lane&15
    const float bias = b2[c0 + lm];
    #pragma unroll
    for (int j = 0; j < 8; ++j) {
        int m = j + half * 8;
        float v = acc[j] + bias;
        H2[(size_t)(r0 + m) * HIDDEN + c0 + lm] = v > 0.0f ? v : 0.0f;
    }
}

// ---------------- mean pool (run-length over sorted batch ids) + classifier --------

__global__ void k_pool_zero(float* __restrict__ sums, float* __restrict__ cnts) {
    int i = blockIdx.x * blockDim.x + threadIdx.x;
    if (i < NUM_GRAPHS * HIDDEN) sums[i] = 0.0f;
    if (i < NUM_GRAPHS) cnts[i] = 0.0f;
}

__global__ void k_pool_scatter(const int* __restrict__ batch, const float* __restrict__ H2,
                               float* __restrict__ sums, float* __restrict__ cnts) {
    __shared__ int gb[POOL_NODES];
    int t  = threadIdx.x;                      // 128 threads = one per feature
    int n0 = blockIdx.x * POOL_NODES;
    int cnt = N_NODES - n0; if (cnt > POOL_NODES) cnt = POOL_NODES;
    if (t < cnt) gb[t] = batch[n0 + t];
    __syncthreads();
    float acc = 0.0f;
    int gprev = gb[0];
    for (int j = 0; j < cnt; ++j) {
        int g = gb[j];
        if (g != gprev) {                      // batch sorted -> few transitions
            atomicAdd(&sums[gprev * HIDDEN + t], acc);
            acc = 0.0f; gprev = g;
        }
        acc += H2[(size_t)(n0 + j) * HIDDEN + t];
    }
    atomicAdd(&sums[gprev * HIDDEN + t], acc);
    if (t == 0) {
        float rc = 0.0f; int gp = gb[0];
        for (int j = 0; j < cnt; ++j) {
            int g = gb[j];
            if (g != gp) { atomicAdd(&cnts[gp], rc); rc = 0.0f; gp = g; }
            rc += 1.0f;
        }
        atomicAdd(&cnts[gp], rc);
    }
}

__global__ void k_cls1(const float* __restrict__ sums, const float* __restrict__ cnts,
                       const float* __restrict__ Wc1, const float* __restrict__ bc1,
                       float* __restrict__ z) {
    int g = blockIdx.x;          // 256
    int o = threadIdx.x;         // 32
    float c = cnts[g];
    float inv = 1.0f / (c > 1.0f ? c : 1.0f);
    float acc = bc1[o];
    #pragma unroll 4
    for (int k = 0; k < HIDDEN; ++k)
        acc += sums[g * HIDDEN + k] * inv * Wc1[k * 32 + o];
    z[g * 32 + o] = acc > 0.0f ? acc : 0.0f;
}

__global__ void k_cls2(const float* __restrict__ z, const float* __restrict__ Wc2,
                       const float* __restrict__ bc2, float* __restrict__ out) {
    int i = blockIdx.x * blockDim.x + threadIdx.x;
    if (i >= NUM_GRAPHS * 2) return;
    int g = i >> 1, c = i & 1;
    float acc = bc2[c];
    #pragma unroll
    for (int j = 0; j < 32; ++j) acc += z[g * 32 + j] * Wc2[j * 2 + c];
    out[i] = 1.0f / (1.0f + expf(-acc));
}

// ---------------- launch ----------------------------------------------------------

extern "C" void kernel_launch(void* const* d_in, const int* in_sizes, int n_in,
                              void* d_out, int out_size, void* d_ws, size_t ws_size,
                              hipStream_t stream) {
    const float* x    = (const float*)d_in[0];
    const int*   ei   = (const int*)  d_in[1];
    const int*   row  = ei;
    const int*   col  = ei + N_EDGES;
    const int*   batch= (const int*)  d_in[2];
    const float* W1   = (const float*)d_in[3];
    const float* b1   = (const float*)d_in[4];
    const float* W2   = (const float*)d_in[5];
    const float* b2   = (const float*)d_in[6];
    const float* Wc1  = (const float*)d_in[7];
    const float* bc1  = (const float*)d_in[8];
    const float* Wc2  = (const float*)d_in[9];
    const float* bc2  = (const float*)d_in[10];
    float* out = (float*)d_out;

    char*  base = (char*)d_ws;
    size_t off  = 0;
    auto carve = [&](size_t bytes) -> char* {
        char* p = base + off;
        off += (bytes + 511) & ~(size_t)511;
        return p;
    };
    float*     deg   = (float*)    carve((size_t)N_NODES * 4);
    float*     dinv  = (float*)    carve((size_t)N_NODES * 4);
    float*     u     = (float*)    carve((size_t)N_NODES * 4);
    float*     tbuf  = (float*)    carve((size_t)N_NODES * 4);
    float*     s1    = (float*)    carve((size_t)N_NODES * 4);
    int*       degi  = (int*)      carve((size_t)N_NODES * 4);
    int*       rowptr= (int*)      carve((size_t)N_NODES * 4);
    int*       cursor= (int*)      carve((size_t)N_NODES * 4);
    int*       bsums = (int*)      carve((size_t)N_SCAN_BLK * 4);
    int*       boff  = (int*)      carve((size_t)N_SCAN_BLK * 4);
    int*       ebuf  = (int*)      carve((size_t)N_EDGES * 4);          // 6.4 MB
    _Float16*  Bt    = (_Float16*) carve((size_t)HIDDEN * HIDDEN * 2);
    float*     sums  = (float*)    carve((size_t)NUM_GRAPHS * HIDDEN * 4);
    float*     cnts  = (float*)    carve((size_t)NUM_GRAPHS * 4);
    float*     z     = (float*)    carve((size_t)NUM_GRAPHS * 32 * 4);
    float*     A     = (float*)    carve((size_t)N_NODES * HIDDEN * 4); // 51.2 MB
    float*     H2    = (float*)    carve((size_t)N_NODES * HIDDEN * 4); // 51.2 MB

    const int TB = 256;
    // degrees + scalar layer-1 aggregation
    k_init_deg  <<<(N_NODES + TB - 1) / TB, TB, 0, stream>>>(deg);
    k_count_deg <<<(N_EDGES + TB - 1) / TB, TB, 0, stream>>>(col, deg);
    k_dinv_u    <<<(N_NODES + TB - 1) / TB, TB, 0, stream>>>(deg, x, dinv, u, tbuf, degi);
    k_scatter_u <<<(N_EDGES + TB - 1) / TB, TB, 0, stream>>>(row, col, u, tbuf);
    k_s1        <<<(N_NODES + TB - 1) / TB, TB, 0, stream>>>(dinv, u, tbuf, s1);

    // CSR build: prefix scan of in-degrees, then bucket edges by target
    k_blocksum   <<<N_SCAN_BLK, SCAN_BLK, 0, stream>>>(degi, bsums);
    k_scan_blocks<<<1, 32, 0, stream>>>(bsums, boff);
    k_scan_local <<<N_SCAN_BLK, SCAN_BLK, 0, stream>>>(degi, boff, rowptr, cursor);
    k_bucket     <<<(N_EDGES + TB - 1) / TB, TB, 0, stream>>>(row, col, cursor, ebuf);

    // atomic-free aggregation of on-the-fly h1 into A
    k_agg_csr    <<<N_NODES / 8, TB, 0, stream>>>(rowptr, degi, ebuf, dinv, s1, W1, b1, A);

    // H2 = relu(A @ W2 + b2) via v_wmma_f32_16x16x32_f16
    k_pack_W2   <<<(HIDDEN * HIDDEN + TB - 1) / TB, TB, 0, stream>>>(W2, Bt);
    k_gemm_wmma <<<N_NODES / 16, TB, 0, stream>>>(A, Bt, b2, H2);

    // mean pool + MLP head
    k_pool_zero   <<<(NUM_GRAPHS * HIDDEN + TB - 1) / TB, TB, 0, stream>>>(sums, cnts);
    k_pool_scatter<<<(N_NODES + POOL_NODES - 1) / POOL_NODES, HIDDEN, 0, stream>>>(batch, H2, sums, cnts);
    k_cls1        <<<NUM_GRAPHS, 32, 0, stream>>>(sums, cnts, Wc1, bc1, z);
    k_cls2        <<<2, TB, 0, stream>>>(z, Wc2, bc2, out);

    (void)in_sizes; (void)n_in; (void)out_size; (void)ws_size;
}
